// LSTM_Autoencoder_10307921510695
// MI455X (gfx1250) — compile-verified, matrix operand order checked by reading, not run on hardware
//
#include <hip/hip_runtime.h>
#include <hip/hip_bf16.h>
#include <cstdint>

// LSTM autoencoder on MI455X (gfx1250).
// f16 weights/activations + v_wmma_f32_16x16x32_f16 (fp32 accum). One fused
// GEMM+LSTM-pointwise kernel per time step; weights converted once and stay
// resident in the 192MB L2 across all 128 steps. Decoder algebra folded:
//   step0 : gates = tanh(cx) @ Whh_d^T + (bd + 0.1*rowsum(Wih_d))
//   step t: gates = h @ (Wih_d + Whh_d)^T + bd
// Data movement uses CDNA5 GLOBAL_LOAD_ASYNC_TO_LDS_B128 (ASYNCcnt) into a
// double-buffered LDS stage: zero VGPR staging (fixes last round's scratch
// spills), DMA of chunk k+1 overlaps the WMMA burst of chunk k, one barrier
// per chunk.

#define Bc 256
#define Tc 64
#define Ic 1024
#define Hc 1024

// LDS row pitch: 40 halves = 80B (16B-aligned rows, odd-ish dword stride).
#define PITCH_H   40
#define PITCH_U32 20
#define A_HALVES      (64 * PITCH_H)           // 2560 halves (5120 B)
#define B_HALVES_LSTM (256 * PITCH_H)          // 10240 halves (20480 B)
#define BUF_HALVES_LSTM (A_HALVES + B_HALVES_LSTM)

typedef __attribute__((ext_vector_type(16))) _Float16 v16h;
typedef __attribute__((ext_vector_type(8)))  float    v8f;

union Frag {
    uint32_t u[8];
    v16h     h;
};

// ---- CDNA5 async global->LDS DMA (no VGPR staging; tracked by ASYNCcnt) ----
__device__ __forceinline__ void async_b128(uint32_t lds_byte_addr, const void* gaddr) {
    asm volatile("global_load_async_to_lds_b128 %0, %1, off"
                 :: "v"(lds_byte_addr), "v"(gaddr) : "memory");
}
__device__ __forceinline__ void wait_async0() {
    asm volatile("s_wait_asynccnt 0x0" ::: "memory");
}

// Gather one 16x32 f16 fragment (A-layout, ISA 7.12.2) from an LDS tile.
// lane 0-15: row M=lane, K pairs {0..7,16..23}; lane 16-31: row M=lane-16,
// K pairs {8..15,24..31}.
__device__ __forceinline__ void load_frag_lds(const uint32_t* lds_u32, int row_base,
                                              Frag& f) {
    const int lane = threadIdx.x & 31;
    const uint32_t* p = lds_u32 + (row_base + (lane & 15)) * PITCH_U32
                        + ((lane >> 4) << 2);
    f.u[0] = p[0];  f.u[1] = p[1];  f.u[2] = p[2];  f.u[3] = p[3];
    f.u[4] = p[8];  f.u[5] = p[9];  f.u[6] = p[10]; f.u[7] = p[11];
}

__device__ __forceinline__ float sigmoidf_(float x) {
    return 1.0f / (1.0f + __expf(-x));
}

// Issue async DMA of one K-chunk (32 halves) of A (64 rows) + B (4 gates x 64
// rows) into the LDS buffer at lds_base (byte address). 5 b128 per thread.
__device__ __forceinline__ void issue_lstm_chunk(const _Float16* __restrict__ A, int lda,
                                                 const _Float16* __restrict__ W, int K,
                                                 int m0, int n0, int k0, int tid,
                                                 uint32_t lds_base) {
    {
        const int arow = tid >> 2, acol = (tid & 3) * 8;
        async_b128(lds_base + (uint32_t)(arow * (PITCH_H * 2) + acol * 2),
                   A + (size_t)(m0 + arow) * lda + k0 + acol);
    }
#pragma unroll
    for (int s = 0; s < 4; ++s) {
        const int idx  = s * 256 + tid;
        const int brow = idx >> 2, bcol = (idx & 3) * 8;
        const int gate = brow >> 6, ncol = brow & 63;
        async_b128(lds_base + (uint32_t)(A_HALVES * 2 + brow * (PITCH_H * 2) + bcol * 2),
                   W + (size_t)(gate * Hc + n0 + ncol) * K + k0 + bcol);
    }
}

// Fused LSTM step: gates[B,4H] = A0@W0^T (+ A1@W1^T) + bias, then pointwise
// update of c (fp32, in-place) and h_out (f16). Grid (4,16), block 256 (8
// wave32). Block tile 64(M) x 64(N over H); wave: wm=wave&3 (16 M rows),
// wn=wave>>2 (32 N cols); acc[gate][ntile].
__global__ __launch_bounds__(256)
void lstm_step_kernel(const _Float16* __restrict__ A0, int lda0,
                      const _Float16* __restrict__ W0, int K0,
                      const _Float16* __restrict__ A1, int lda1,
                      const _Float16* __restrict__ W1, int K1,
                      const float* __restrict__ bias,
                      float* __restrict__ c,
                      _Float16* __restrict__ h_out) {
    __shared__ __align__(16) _Float16 lds[2][BUF_HALVES_LSTM];   // 2 x 25.6 KB

    const int tid  = threadIdx.x;
    const int wave = tid >> 5;
    const int wm   = wave & 3;
    const int wn   = wave >> 2;
    const int m0   = blockIdx.x * 64;
    const int n0   = blockIdx.y * 64;

    const uint32_t ldsb[2] = { (uint32_t)(uintptr_t)&lds[0][0],
                               (uint32_t)(uintptr_t)&lds[1][0] };

    v8f acc[4][2] = {};

    for (int pass = 0; pass < 2; ++pass) {
        const _Float16* A = pass ? A1 : A0;
        if (A == nullptr) break;
        const _Float16* W = pass ? W1 : W0;
        const int lda = pass ? lda1 : lda0;
        const int K   = pass ? K1 : K0;

        issue_lstm_chunk(A, lda, W, K, m0, n0, 0, tid, ldsb[0]);
        wait_async0();
        __syncthreads();

        for (int k0 = 0; k0 < K; k0 += 32) {
            const int cur = (k0 >> 5) & 1;
            if (k0 + 32 < K)               // DMA of next chunk overlaps compute
                issue_lstm_chunk(A, lda, W, K, m0, n0, k0 + 32, tid, ldsb[cur ^ 1]);

            const _Float16* bufc = &lds[cur][0];
            const uint32_t* a32 = (const uint32_t*)bufc;
            const uint32_t* b32 = (const uint32_t*)(bufc + A_HALVES);

            Frag a;
            load_frag_lds(a32, wm * 16, a);
            // Two bursts: preload 4 B-fragments, then 4 back-to-back WMMAs.
#pragma unroll
            for (int gp = 0; gp < 2; ++gp) {
                Frag b[4];
#pragma unroll
                for (int i = 0; i < 4; ++i) {
                    const int g = gp * 2 + (i >> 1), t = i & 1;
                    load_frag_lds(b32, g * 64 + wn * 32 + t * 16, b[i]);
                }
#pragma unroll
                for (int i = 0; i < 4; ++i) {
                    const int g = gp * 2 + (i >> 1), t = i & 1;
                    acc[g][t] = __builtin_amdgcn_wmma_f32_16x16x32_f16(
                        false, a.h, false, b[i].h, (short)0, acc[g][t], false, false);
                }
            }

            wait_async0();        // our DMA into the other buffer has landed
            __syncthreads();      // everyone done reading cur / writing nxt
        }
    }

    // Pointwise LSTM epilogue. C/D layout: lane 0-15 -> N=lane, M=r; lane
    // 16-31 -> N=lane-16, M=r+8.
    const int lane   = tid & 31;
    const int half   = lane >> 4;
    const int nl     = lane & 15;
    const int m_base = m0 + wm * 16 + half * 8;
#pragma unroll
    for (int t = 0; t < 2; ++t) {
        const int n = n0 + wn * 32 + t * 16 + nl;
        const float bi = bias[n];
        const float bf = bias[Hc + n];
        const float bg = bias[2 * Hc + n];
        const float bo = bias[3 * Hc + n];
#pragma unroll
        for (int r = 0; r < 8; ++r) {
            const int m = m_base + r;
            const size_t off = (size_t)m * Hc + n;
            const float iv = sigmoidf_(acc[0][t][r] + bi);
            const float fv = sigmoidf_(acc[1][t][r] + bf);
            const float gv = tanhf(acc[2][t][r] + bg);
            const float ov = sigmoidf_(acc[3][t][r] + bo);
            const float cn = fv * c[off] + iv * gv;
            c[off] = cn;
            h_out[off] = (_Float16)(ov * tanhf(cn));
        }
    }
}

// z = tanh(h @ Wz^T + bz) written into out[b, tstep, :] (fp32).
__global__ __launch_bounds__(256)
void zproj_kernel(const _Float16* __restrict__ hbuf,
                  const _Float16* __restrict__ Wzp,
                  const float* __restrict__ bz,
                  float* __restrict__ out, int tstep) {
    __shared__ __align__(16) _Float16 lds[2][2 * A_HALVES];   // 2 x 10.2 KB

    const int tid  = threadIdx.x;
    const int wave = tid >> 5;
    const int wm   = wave & 3;
    const int wn   = wave >> 2;
    const int m0   = blockIdx.x * 64;
    const int n0   = blockIdx.y * 64;

    const uint32_t ldsb[2] = { (uint32_t)(uintptr_t)&lds[0][0],
                               (uint32_t)(uintptr_t)&lds[1][0] };

    v8f acc[2] = {};

    auto issue = [&](int k0, uint32_t base) {
        const int row = tid >> 2, col = (tid & 3) * 8;
        async_b128(base + (uint32_t)(row * (PITCH_H * 2) + col * 2),
                   hbuf + (size_t)(m0 + row) * Hc + k0 + col);
        async_b128(base + (uint32_t)(A_HALVES * 2 + row * (PITCH_H * 2) + col * 2),
                   Wzp + (size_t)(n0 + row) * Hc + k0 + col);
    };

    issue(0, ldsb[0]);
    wait_async0();
    __syncthreads();

    for (int k0 = 0; k0 < Hc; k0 += 32) {
        const int cur = (k0 >> 5) & 1;
        if (k0 + 32 < Hc) issue(k0 + 32, ldsb[cur ^ 1]);

        const _Float16* bufc = &lds[cur][0];
        const uint32_t* a32 = (const uint32_t*)bufc;
        const uint32_t* b32 = (const uint32_t*)(bufc + A_HALVES);

        Frag a;
        load_frag_lds(a32, wm * 16, a);
        Frag b[2];
        load_frag_lds(b32, wn * 32,      b[0]);
        load_frag_lds(b32, wn * 32 + 16, b[1]);
#pragma unroll
        for (int t = 0; t < 2; ++t)
            acc[t] = __builtin_amdgcn_wmma_f32_16x16x32_f16(
                false, a.h, false, b[t].h, (short)0, acc[t], false, false);

        wait_async0();
        __syncthreads();
    }

    const int lane   = tid & 31;
    const int half   = lane >> 4;
    const int nl     = lane & 15;
    const int m_base = m0 + wm * 16 + half * 8;
#pragma unroll
    for (int t = 0; t < 2; ++t) {
        const int n = n0 + wn * 32 + t * 16 + nl;
        const float b = bz[n];
#pragma unroll
        for (int r = 0; r < 8; ++r) {
            const int m = m_base + r;
            out[((size_t)m * Tc + tstep) * Ic + n] = tanhf(acc[t][r] + b);
        }
    }
}

// ---- prep / pointwise helper kernels ----

__global__ void cvt_f32_f16(const float* __restrict__ s, _Float16* __restrict__ d, int n) {
    int i = blockIdx.x * blockDim.x + threadIdx.x;
    if (i < n) d[i] = (_Float16)s[i];
}

__global__ void add_cvt_f16(const float* __restrict__ a, const float* __restrict__ b,
                            _Float16* __restrict__ d, int n) {
    int i = blockIdx.x * blockDim.x + threadIdx.x;
    if (i < n) d[i] = (_Float16)(a[i] + b[i]);
}

__global__ void add_f32(const float* __restrict__ a, const float* __restrict__ b,
                        float* __restrict__ d, int n) {
    int i = blockIdx.x * blockDim.x + threadIdx.x;
    if (i < n) d[i] = a[i] + b[i];
}

// bd0[j] = bih_d[j] + bhh_d[j] + 0.1 * sum_k Wih_d[j,k]  (decoder step-0 const input)
__global__ void bd0_kernel(const float* __restrict__ bih, const float* __restrict__ bhh,
                           const float* __restrict__ Wih, float* __restrict__ bd0) {
    int j = blockIdx.x * blockDim.x + threadIdx.x;
    if (j < 4 * Hc) {
        float s = 0.f;
        for (int k = 0; k < Hc; ++k) s += Wih[(size_t)j * Hc + k];
        bd0[j] = bih[j] + bhh[j] + 0.1f * s;
    }
}

__global__ void init_state(_Float16* __restrict__ h, float* __restrict__ c, int n) {
    int i = blockIdx.x * blockDim.x + threadIdx.x;
    if (i < n) { h[i] = (_Float16)0.1f; c[i] = 0.1f; }
}

__global__ void tanh_c_to_h(const float* __restrict__ c, _Float16* __restrict__ h, int n) {
    int i = blockIdx.x * blockDim.x + threadIdx.x;
    if (i < n) h[i] = (_Float16)tanhf(c[i]);
}

extern "C" void kernel_launch(void* const* d_in, const int* in_sizes, int n_in,
                              void* d_out, int out_size, void* d_ws, size_t ws_size,
                              hipStream_t stream) {
    (void)in_sizes; (void)n_in; (void)out_size; (void)ws_size;
    const float* x     = (const float*)d_in[0];
    const float* Wih_e = (const float*)d_in[1];
    const float* Whh_e = (const float*)d_in[2];
    const float* bih_e = (const float*)d_in[3];
    const float* bhh_e = (const float*)d_in[4];
    const float* Wih_d = (const float*)d_in[5];
    const float* Whh_d = (const float*)d_in[6];
    const float* bih_d = (const float*)d_in[7];
    const float* bhh_d = (const float*)d_in[8];
    const float* Wz    = (const float*)d_in[9];
    const float* bz    = (const float*)d_in[10];
    float* out = (float*)d_out;

    char* ws = (char*)d_ws;
    size_t off = 0;
    auto carve = [&](size_t bytes) -> void* {
        void* p = ws + off;
        off += (bytes + 255) & ~(size_t)255;
        return p;
    };
    _Float16* x16     = (_Float16*)carve((size_t)Bc * Tc * Ic * 2);  // 32 MB
    _Float16* Wih_e16 = (_Float16*)carve((size_t)4 * Hc * Ic * 2);   // 8 MB
    _Float16* Whh_e16 = (_Float16*)carve((size_t)4 * Hc * Hc * 2);
    _Float16* Whh_d16 = (_Float16*)carve((size_t)4 * Hc * Hc * 2);
    _Float16* Wsum16  = (_Float16*)carve((size_t)4 * Hc * Hc * 2);
    _Float16* Wz16    = (_Float16*)carve((size_t)Ic * Hc * 2);
    float*    be      = (float*)carve((size_t)4 * Hc * 4);
    float*    bd      = (float*)carve((size_t)4 * Hc * 4);
    float*    bd0     = (float*)carve((size_t)4 * Hc * 4);
    _Float16* hA      = (_Float16*)carve((size_t)Bc * Hc * 2);
    _Float16* hB      = (_Float16*)carve((size_t)Bc * Hc * 2);
    float*    cbuf    = (float*)carve((size_t)Bc * Hc * 4);

    auto nb = [](size_t n) { return (unsigned)((n + 255) / 256); };

    // One-time (per call) weight/input conversion; weights then live in L2.
    cvt_f32_f16<<<nb((size_t)Bc * Tc * Ic), 256, 0, stream>>>(x, x16, Bc * Tc * Ic);
    cvt_f32_f16<<<nb((size_t)4 * Hc * Ic), 256, 0, stream>>>(Wih_e, Wih_e16, 4 * Hc * Ic);
    cvt_f32_f16<<<nb((size_t)4 * Hc * Hc), 256, 0, stream>>>(Whh_e, Whh_e16, 4 * Hc * Hc);
    cvt_f32_f16<<<nb((size_t)4 * Hc * Hc), 256, 0, stream>>>(Whh_d, Whh_d16, 4 * Hc * Hc);
    add_cvt_f16<<<nb((size_t)4 * Hc * Hc), 256, 0, stream>>>(Wih_d, Whh_d, Wsum16, 4 * Hc * Hc);
    cvt_f32_f16<<<nb((size_t)Ic * Hc), 256, 0, stream>>>(Wz, Wz16, Ic * Hc);
    add_f32<<<nb(4 * Hc), 256, 0, stream>>>(bih_e, bhh_e, be, 4 * Hc);
    add_f32<<<nb(4 * Hc), 256, 0, stream>>>(bih_d, bhh_d, bd, 4 * Hc);
    bd0_kernel<<<nb(4 * Hc), 256, 0, stream>>>(bih_d, bhh_d, Wih_d, bd0);
    init_state<<<nb((size_t)Bc * Hc), 256, 0, stream>>>(hA, cbuf, Bc * Hc);

    const dim3 grid(Bc / 64, Hc / 64), blk(256);
    _Float16* hin = hA;
    _Float16* hout = hB;

    // Encoder: gates = x_t @ Wih_e^T + h @ Whh_e^T + be
    for (int t = 0; t < Tc; ++t) {
        lstm_step_kernel<<<grid, blk, 0, stream>>>(
            x16 + (size_t)t * Ic, Tc * Ic, Wih_e16, Ic,
            hin, Hc, Whh_e16, Hc, be, cbuf, hout);
        _Float16* tmp = hin; hin = hout; hout = tmp;
    }

    // Decoder init: h = tanh(cx) (encoder hx is dead); c = cx unchanged.
    tanh_c_to_h<<<nb((size_t)Bc * Hc), 256, 0, stream>>>(cbuf, hin, Bc * Hc);

    // Decoder step 0: gates = tanh(cx) @ Whh_d^T + bd0  (0.1-input folded into bd0)
    lstm_step_kernel<<<grid, blk, 0, stream>>>(
        hin, Hc, Whh_d16, Hc, nullptr, 0, nullptr, 0, bd0, cbuf, hout);
    zproj_kernel<<<grid, blk, 0, stream>>>(hout, Wz16, bz, out, 0);
    { _Float16* tmp = hin; hin = hout; hout = tmp; }

    // Decoder steps 1..T-1: gates = h @ (Wih_d+Whh_d)^T + bd
    for (int t = 1; t < Tc; ++t) {
        lstm_step_kernel<<<grid, blk, 0, stream>>>(
            hin, Hc, Wsum16, Hc, nullptr, 0, nullptr, 0, bd, cbuf, hout);
        zproj_kernel<<<grid, blk, 0, stream>>>(hout, Wz16, bz, out, t);
        _Float16* tmp = hin; hin = hout; hout = tmp;
    }
}
